// PGCN_72258529788421
// MI455X (gfx1250) — matrix operand order repaired and checked by reading, not compiled, not run on hardware
//
#include <hip/hip_runtime.h>
#include <hip/hip_bf16.h>
#include <stdint.h>

// Problem constants (match reference)
#define N_NODES 50000
#define N_EDGES 800000
#define IN_CH   1280
#define HIDDEN  256
#define NUM_G   64

typedef __bf16 v16bf __attribute__((ext_vector_type(16)));
typedef float  v8f   __attribute__((ext_vector_type(8)));

union BFrag {
    unsigned short u[16];
    uint4          q[2];
    v16bf          v;
};

__device__ __forceinline__ unsigned short f2bf_rne(float f) {
    unsigned u = __float_as_uint(f);
    unsigned r = u + 0x7FFFu + ((u >> 16) & 1u);
    return (unsigned short)(r >> 16);
}
__device__ __forceinline__ float bf2f(unsigned short h) {
    return __uint_as_float(((unsigned)h) << 16);
}
__device__ __forceinline__ void split_bf(float f, unsigned short& hi, unsigned short& lo) {
    hi = f2bf_rne(f);
    lo = f2bf_rne(f - bf2f(hi));
}
// pack two floats' bf16(RNE) into one dword (a -> low 16, b -> high 16)
__device__ __forceinline__ unsigned pack2_rne(float a, float b) {
    unsigned ua = __float_as_uint(a), ub = __float_as_uint(b);
    unsigned ra = ua + 0x7FFFu + ((ua >> 16) & 1u);
    unsigned rb = ub + 0x7FFFu + ((ub >> 16) & 1u);
    return (ra >> 16) | (rb & 0xFFFF0000u);
}

// ---------------------------------------------------------------------------
// 1) init: zero workspace accumulators
// ---------------------------------------------------------------------------
__global__ void k_init(int* degN, float* gsum, int* gcnt) {
    int i = blockIdx.x * blockDim.x + threadIdx.x;
    if (i < N_NODES) degN[i] = 0;
    if (i < NUM_G * HIDDEN) gsum[i] = 0.0f;
    if (i < NUM_G) gcnt[i] = 0;
}

// ---------------------------------------------------------------------------
// 2) pre-pack W into WMMA B-fragment layout (hi/lo bf16 split).
//    B (32x16, bf16): lane = h*16 + (n&15); element e -> K = 16*h + e.
//    Fragment f = kt*16 + ntile stored contiguously: 32 lanes x 16 bf16.
// ---------------------------------------------------------------------------
__global__ void k_wprep(const float* __restrict__ W,
                        unsigned short* __restrict__ wfh,
                        unsigned short* __restrict__ wfl) {
    int tid = blockIdx.x * blockDim.x + threadIdx.x;   // 40*16*32 = 20480
    if (tid >= 40 * 16 * 32) return;
    int lane  = tid & 31;
    int ntile = (tid >> 5) & 15;
    int kt    = tid >> 9;
    int h     = lane >> 4;
    int n     = ntile * 16 + (lane & 15);
    size_t base = (size_t)tid * 16;
#pragma unroll
    for (int e = 0; e < 16; ++e) {
        int k = kt * 32 + h * 16 + e;
        float f = W[(size_t)k * HIDDEN + n];
        unsigned short hi, lo;
        split_bf(f, hi, lo);
        wfh[base + e] = hi;
        wfl[base + e] = lo;
    }
}

// ---------------------------------------------------------------------------
// 3) GEMM: xw = x @ W via bf16x3 WMMA (fp32-accurate).
//    Block: 256 threads = 8 waves, 2(M)x4(N); block tile 64 rows x 256 cols.
//    Wave tile: 32 rows x 64 cols = 2 M-subtiles x 4 N-tiles.
//    Staging converts fp32 -> bf16 hi/lo ONCE per block into two LDS tiles;
//    fragment build is then pure ds_load_b128 (A-layout K-runs are contiguous).
// ---------------------------------------------------------------------------
__global__ __launch_bounds__(256) void k_gemm(const float* __restrict__ x,
                                              const unsigned short* __restrict__ wfh,
                                              const unsigned short* __restrict__ wfl,
                                              float* __restrict__ xw) {
    // 64 rows x 32 K of bf16; row stride 40 ushorts (80B, 16B-aligned,
    // 20-dword stride keeps the 16-lane b128 read pattern bank-clean)
    __shared__ __attribute__((aligned(16))) unsigned short sh_hi[64][40];
    __shared__ __attribute__((aligned(16))) unsigned short sh_lo[64][40];

    const int tid  = threadIdx.x;
    const int lane = tid & 31;
    const int wave = tid >> 5;
    const int mw   = wave >> 2;    // 0..1
    const int nw   = wave & 3;     // 0..3
    const int m0   = blockIdx.x * 64;
    const int lr   = lane & 15;
    const int h    = lane >> 4;

    v8f acc[2][4];
#pragma unroll
    for (int ms = 0; ms < 2; ++ms)
#pragma unroll
        for (int nt = 0; nt < 4; ++nt)
            acc[ms][nt] = (v8f){0.f,0.f,0.f,0.f,0.f,0.f,0.f,0.f};

    for (int kt = 0; kt < IN_CH / 32; ++kt) {
        const int k0 = kt * 32;
        // stage + convert: each thread handles 8 consecutive floats of one row
        {
            int r  = tid >> 2;
            int cq = (tid & 3) * 8;      // 0,8,16,24
            int row = m0 + r;
            float4 a = {0.f,0.f,0.f,0.f}, b = {0.f,0.f,0.f,0.f};
            if (row < N_NODES) {
                const float4* p = reinterpret_cast<const float4*>(
                    x + (size_t)row * IN_CH + k0 + cq);
                a = p[0]; b = p[1];
            }
            float f[8] = {a.x, a.y, a.z, a.w, b.x, b.y, b.z, b.w};
            uint4 hq, lq;
            unsigned* hp = reinterpret_cast<unsigned*>(&hq);
            unsigned* lp = reinterpret_cast<unsigned*>(&lq);
#pragma unroll
            for (int j = 0; j < 4; ++j) {
                float f0 = f[2 * j], f1 = f[2 * j + 1];
                unsigned hpk = pack2_rne(f0, f1);
                hp[j] = hpk;
                float h0 = bf2f((unsigned short)(hpk & 0xFFFFu));
                float h1 = bf2f((unsigned short)(hpk >> 16));
                lp[j] = pack2_rne(f0 - h0, f1 - h1);
            }
            *reinterpret_cast<uint4*>(&sh_hi[r][cq]) = hq;
            *reinterpret_cast<uint4*>(&sh_lo[r][cq]) = lq;
        }
        __syncthreads();

        // A fragments (16x32 bf16): lane-half h, elem e -> K = 16h+e
        // -> two contiguous 8-bf16 runs at K = 16h and K = 16h+8.
        BFrag ah[2], al[2];
#pragma unroll
        for (int ms = 0; ms < 2; ++ms) {
            int r = 32 * mw + 16 * ms + lr;
            ah[ms].q[0] = *reinterpret_cast<const uint4*>(&sh_hi[r][16 * h]);
            ah[ms].q[1] = *reinterpret_cast<const uint4*>(&sh_hi[r][16 * h + 8]);
            al[ms].q[0] = *reinterpret_cast<const uint4*>(&sh_lo[r][16 * h]);
            al[ms].q[1] = *reinterpret_cast<const uint4*>(&sh_lo[r][16 * h + 8]);
        }

#pragma unroll
        for (int nt = 0; nt < 4; ++nt) {
            int ntg = nw * 4 + nt;
            size_t boff = ((size_t)(kt * 16 + ntg) * 32 + lane) * 16;
            BFrag bh, bl;
            const uint4* ph = reinterpret_cast<const uint4*>(wfh + boff);
            const uint4* pl = reinterpret_cast<const uint4*>(wfl + boff);
            bh.q[0] = ph[0]; bh.q[1] = ph[1];
            bl.q[0] = pl[0]; bl.q[1] = pl[1];
#pragma unroll
            for (int ms = 0; ms < 2; ++ms) {
                acc[ms][nt] = __builtin_amdgcn_wmma_f32_16x16x32_bf16(
                    false, ah[ms].v, false, bh.v, (short)0, acc[ms][nt], false, false);
                acc[ms][nt] = __builtin_amdgcn_wmma_f32_16x16x32_bf16(
                    false, ah[ms].v, false, bl.v, (short)0, acc[ms][nt], false, false);
                acc[ms][nt] = __builtin_amdgcn_wmma_f32_16x16x32_bf16(
                    false, al[ms].v, false, bh.v, (short)0, acc[ms][nt], false, false);
            }
        }
        __syncthreads();
    }

    // store: C/D layout -> lane l, VGPR v: M = v + 8*(l>>4), N = l&15
#pragma unroll
    for (int ms = 0; ms < 2; ++ms)
#pragma unroll
        for (int nt = 0; nt < 4; ++nt)
#pragma unroll
            for (int v = 0; v < 8; ++v) {
                int m = m0 + 32 * mw + 16 * ms + v + 8 * h;
                int n = nw * 64 + nt * 16 + lr;
                if (m < N_NODES) xw[(size_t)m * HIDDEN + n] = acc[ms][nt][v];
            }
}

// ---------------------------------------------------------------------------
// 4) degree histogram (in-degree of col, excluding self loop) + per-graph counts
// ---------------------------------------------------------------------------
__global__ void k_degree(const int* __restrict__ ei, const int* __restrict__ batch,
                         int* degN, int* gcnt) {
    int i = blockIdx.x * blockDim.x + threadIdx.x;
    if (i < N_EDGES) atomicAdd(&degN[ei[N_EDGES + i]], 1);   // col = targets
    if (i < N_NODES) atomicAdd(&gcnt[batch[i]], 1);
}

// ---------------------------------------------------------------------------
// 5) single-workgroup exclusive scan of degN -> start/cursor; dis = rsqrt(deg+1)
// ---------------------------------------------------------------------------
__global__ __launch_bounds__(1024) void k_scan(const int* __restrict__ degN,
                                               float* __restrict__ dis,
                                               int* __restrict__ start,
                                               int* __restrict__ cursor) {
    __shared__ int s[1024];
    __shared__ int running;
    int tid = threadIdx.x;
    if (tid == 0) running = 0;
    __syncthreads();
    for (int base = 0; base < N_NODES; base += 1024) {
        int i = base + tid;
        int v = (i < N_NODES) ? degN[i] : 0;
        if (i < N_NODES) dis[i] = rsqrtf((float)(v + 1));  // self loop included
        s[tid] = v;
        __syncthreads();
        for (int off = 1; off < 1024; off <<= 1) {
            int t = (tid >= off) ? s[tid - off] : 0;
            __syncthreads();
            s[tid] += t;
            __syncthreads();
        }
        int excl = s[tid] - v;
        int run = running;
        if (i < N_NODES) { start[i] = run + excl; cursor[i] = run + excl; }
        int tot = s[1023];
        __syncthreads();
        if (tid == 0) running = run + tot;
        __syncthreads();
    }
}

// ---------------------------------------------------------------------------
// 6) CSR bucket fill: bucket[slot] = src for each in-edge of dst
// ---------------------------------------------------------------------------
__global__ void k_fill(const int* __restrict__ ei, int* cursor, int* __restrict__ bucket) {
    int i = blockIdx.x * blockDim.x + threadIdx.x;
    if (i >= N_EDGES) return;
    int dst  = ei[N_EDGES + i];
    int slot = atomicAdd(&cursor[dst], 1);
    bucket[slot] = ei[i];
}

// ---------------------------------------------------------------------------
// 7) per-node gather-aggregate + bias + ReLU + fused pooling accumulation.
//    One wave32 per node; lane owns 8 contiguous channels (32B coalesced).
// ---------------------------------------------------------------------------
__global__ __launch_bounds__(256) void k_aggregate(const float* __restrict__ xw,
                                                   const float* __restrict__ dis,
                                                   const int* __restrict__ start,
                                                   const int* __restrict__ degN,
                                                   const int* __restrict__ bucket,
                                                   const int* __restrict__ batch,
                                                   const float* __restrict__ bias,
                                                   float* gsum) {
    int wave = threadIdx.x >> 5;
    int lane = threadIdx.x & 31;
    int node = blockIdx.x * 8 + wave;
    if (node >= N_NODES) return;

    float di = dis[node];
    int c0 = lane * 8;
    const float4* self = reinterpret_cast<const float4*>(xw + (size_t)node * HIDDEN + c0);
    float wself = di * di;
    float4 a0 = self[0], a1 = self[1];
    float4 acc0 = {wself * a0.x, wself * a0.y, wself * a0.z, wself * a0.w};
    float4 acc1 = {wself * a1.x, wself * a1.y, wself * a1.z, wself * a1.w};

    int s0  = start[node];
    int cnt = degN[node];
    int src = 0; float dsrc = 0.f;
    if (cnt > 0) { src = bucket[s0]; dsrc = dis[src]; }
    for (int j = 0; j < cnt; ++j) {
        int nsrc = 0; float ndsrc = 0.f;
        if (j + 1 < cnt) { nsrc = bucket[s0 + j + 1]; ndsrc = dis[nsrc]; }
        float wgt = di * dsrc;
        const float4* p = reinterpret_cast<const float4*>(xw + (size_t)src * HIDDEN + c0);
        float4 v0 = p[0], v1 = p[1];
        acc0.x += wgt * v0.x; acc0.y += wgt * v0.y; acc0.z += wgt * v0.z; acc0.w += wgt * v0.w;
        acc1.x += wgt * v1.x; acc1.y += wgt * v1.y; acc1.z += wgt * v1.z; acc1.w += wgt * v1.w;
        src = nsrc; dsrc = ndsrc;
    }

    int g = batch[node];
    float* gs = gsum + (size_t)g * HIDDEN + c0;
    float h;
    h = fmaxf(acc0.x + bias[c0 + 0], 0.f); atomicAdd(&gs[0], h);
    h = fmaxf(acc0.y + bias[c0 + 1], 0.f); atomicAdd(&gs[1], h);
    h = fmaxf(acc0.z + bias[c0 + 2], 0.f); atomicAdd(&gs[2], h);
    h = fmaxf(acc0.w + bias[c0 + 3], 0.f); atomicAdd(&gs[3], h);
    h = fmaxf(acc1.x + bias[c0 + 4], 0.f); atomicAdd(&gs[4], h);
    h = fmaxf(acc1.y + bias[c0 + 5], 0.f); atomicAdd(&gs[5], h);
    h = fmaxf(acc1.z + bias[c0 + 6], 0.f); atomicAdd(&gs[6], h);
    h = fmaxf(acc1.w + bias[c0 + 7], 0.f); atomicAdd(&gs[7], h);
}

// ---------------------------------------------------------------------------
// 8) finalize: mean pool
// ---------------------------------------------------------------------------
__global__ void k_final(const float* __restrict__ gsum, const int* __restrict__ gcnt,
                        float* __restrict__ out) {
    int i = blockIdx.x * blockDim.x + threadIdx.x;
    if (i >= NUM_G * HIDDEN) return;
    float c = fmaxf((float)gcnt[i / HIDDEN], 1.0f);
    out[i] = gsum[i] / c;
}

// ---------------------------------------------------------------------------
extern "C" void kernel_launch(void* const* d_in, const int* in_sizes, int n_in,
                              void* d_out, int out_size, void* d_ws, size_t ws_size,
                              hipStream_t stream) {
    const float* x     = (const float*)d_in[0];
    const int*   ei    = (const int*)d_in[1];
    const int*   batch = (const int*)d_in[2];
    const float* W     = (const float*)d_in[3];
    const float* bias  = (const float*)d_in[4];
    float* out = (float*)d_out;

    // carve workspace (aligned to 256B)
    char* p = (char*)d_ws;
    auto take = [&](size_t bytes) {
        char* r = p;
        p += (bytes + 255) & ~(size_t)255;
        return r;
    };
    float*          xw     = (float*)take((size_t)N_NODES * HIDDEN * 4);
    unsigned short* wfh    = (unsigned short*)take((size_t)40 * 16 * 32 * 16 * 2);
    unsigned short* wfl    = (unsigned short*)take((size_t)40 * 16 * 32 * 16 * 2);
    int*            degN   = (int*)take((size_t)N_NODES * 4);
    float*          dis    = (float*)take((size_t)N_NODES * 4);
    int*            start  = (int*)take((size_t)N_NODES * 4);
    int*            cursor = (int*)take((size_t)N_NODES * 4);
    int*            bucket = (int*)take((size_t)N_EDGES * 4);
    float*          gsum   = (float*)take((size_t)NUM_G * HIDDEN * 4);
    int*            gcnt   = (int*)take((size_t)NUM_G * 4);

    k_init<<<(N_NODES + 255) / 256, 256, 0, stream>>>(degN, gsum, gcnt);
    k_wprep<<<(40 * 16 * 32 + 255) / 256, 256, 0, stream>>>(W, wfh, wfl);
    k_gemm<<<(N_NODES + 63) / 64, 256, 0, stream>>>(x, wfh, wfl, xw);
    k_degree<<<(N_EDGES + 255) / 256, 256, 0, stream>>>(ei, batch, degN, gcnt);
    k_scan<<<1, 1024, 0, stream>>>(degN, dis, start, cursor);
    k_fill<<<(N_EDGES + 255) / 256, 256, 0, stream>>>(ei, cursor, bucket);
    k_aggregate<<<(N_NODES + 7) / 8, 256, 0, stream>>>(xw, dis, start, degN, bucket,
                                                       batch, bias, gsum);
    k_final<<<(NUM_G * HIDDEN + 255) / 256, 256, 0, stream>>>(gsum, gcnt, out);
}
